// RecurrentVecEncoder_28054726378297
// MI455X (gfx1250) — matrix operand order repaired and checked by reading, not compiled, not run on hardware
//
#include <hip/hip_runtime.h>

// Problem constants (match reference)
#define BB 64
#define TT 2048
#define FF 256
#define HH 256
#define GG (4 * HH) // 1024

typedef __attribute__((ext_vector_type(16))) __bf16 bf16x16;
typedef __attribute__((ext_vector_type(8)))  __bf16 bf16x8;
typedef __attribute__((ext_vector_type(4)))  __bf16 bf16x4;
typedef __attribute__((ext_vector_type(8)))  float  f32x8;

union Frag {
    bf16x16 v;
    struct Halves { bf16x8 lo, hi; } p;
};

// ---- fragment loaders -------------------------------------------------------
// A-matrix 16x32 bf16 (ISA 7.12.2): lanes 0-15 -> M=lane, K={0..7,16..23};
// lanes 16-31 -> M=lane-16, K={8..15,24..31}. Row-major tile, ldk = 256.
__device__ __forceinline__ bf16x16 load_a_frag(const __bf16* tile, int lane, int s) {
    int m  = lane & 15;
    int kb = s * 32 + (lane >> 4) * 8;
    Frag f;
    f.p.lo = *(const bf16x8*)(tile + m * 256 + kb);
    f.p.hi = *(const bf16x8*)(tile + m * 256 + kb + 16);
    return f.v;
}

// B-matrix 32x16 bf16: lanes 0-15 -> N=lane, K={0..7,16..23}; lanes 16-31 -> N=lane-16,
// K={8..15,24..31}. Weights stored row-major [N][K=256].
__device__ __forceinline__ bf16x16 load_b_frag(const __bf16* W, int nbase, int lane, int s) {
    int n  = nbase + (lane & 15);
    int kb = s * 32 + (lane >> 4) * 8;
    Frag f;
    f.p.lo = *(const bf16x8*)(W + n * 256 + kb);
    f.p.hi = *(const bf16x8*)(W + n * 256 + kb + 16);
    return f.v;
}

__device__ __forceinline__ f32x8 wmma_bf16(bf16x16 a, bf16x16 b, f32x8 c) {
    return __builtin_amdgcn_wmma_f32_16x16x32_bf16(false, a, false, b, (short)0, c, false, false);
}

__device__ __forceinline__ unsigned short f2bf_rne(float x) {
    unsigned u = __float_as_uint(x);
    unsigned r = (u + 0x7FFFu + ((u >> 16) & 1u)) >> 16;
    return (unsigned short)r;
}

__device__ __forceinline__ float sigmf(float x) { return 1.0f / (1.0f + __expf(-x)); }

// ---- K0: weight conversion + bias fold --------------------------------------
__global__ void prep_kernel(const float* __restrict__ Wenc, const float* __restrict__ Wih,
                            const float* __restrict__ Whh, const float* __restrict__ bih,
                            const float* __restrict__ bhh,
                            unsigned short* __restrict__ wencb, unsigned short* __restrict__ wihb,
                            unsigned short* __restrict__ whhb, float* __restrict__ bias_sum) {
    int i = blockIdx.x * blockDim.x + threadIdx.x;
    if (i < HH * FF) wencb[i] = f2bf_rne(Wenc[i]);
    if (i < GG * HH) {
        wihb[i] = f2bf_rne(Wih[i]);
        whhb[i] = f2bf_rne(Whh[i]);
    }
    if (i < GG) bias_sum[i] = bih[i] + bhh[i];
}

// ---- K1: fused encoder GEMM + input-gate GEMM -------------------------------
// M-block = 64 rows (4 M-subtiles of 16) per 256-thread workgroup. Per K-step
// all operands (4 A-frags + 2 B-frags) are materialized in distinct registers
// BEFORE the burst of 8 independent WMMAs, so ds_load latency overlaps the
// matrix pipe instead of serializing it.
__global__ __launch_bounds__(256) void encoder_kernel(
    const float* __restrict__ h, const float* __restrict__ b_enc,
    const __bf16* __restrict__ Wenc, const __bf16* __restrict__ Wih,
    const float* __restrict__ bias_sum, float* __restrict__ xg) {
    extern __shared__ unsigned char smem[];
    __bf16* Atile = (__bf16*)smem;                // 64x256 bf16 = 32 KB (staged h)
    __bf16* Xtile = (__bf16*)(smem + 32 * 1024);  // 64x256 bf16 = 32 KB (relu out)

    const int tid  = threadIdx.x;
    const int lane = tid & 31;
    const int wave = tid >> 5; // 0..7
    const long row0 = (long)blockIdx.x * 64; // flat row = b*T + t

    // Stage 64x256 f32 -> bf16 into LDS (coalesced)
#pragma unroll
    for (int i = 0; i < 64; ++i) {
        int e = i * 256 + tid;
        int m = e >> 8, c = e & 255;
        Atile[m * 256 + c] = (__bf16)h[(row0 + m) * FF + c];
    }
    __syncthreads();

    // GEMM1: x = relu(h * W_enc^T + b_enc); wave owns 2 N-tiles x 4 M-subtiles
    {
        f32x8 acc[2][4] = {};
        const int nb0 = (wave * 2 + 0) * 16;
        const int nb1 = (wave * 2 + 1) * 16;
#pragma unroll
        for (int s = 0; s < 8; ++s) {
            bf16x16 af[4];
#pragma unroll
            for (int mt = 0; mt < 4; ++mt)
                af[mt] = load_a_frag(Atile + mt * 16 * 256, lane, s);
            bf16x16 b0 = load_b_frag(Wenc, nb0, lane, s);
            bf16x16 b1 = load_b_frag(Wenc, nb1, lane, s);
#pragma unroll
            for (int mt = 0; mt < 4; ++mt) acc[0][mt] = wmma_bf16(af[mt], b0, acc[0][mt]);
#pragma unroll
            for (int mt = 0; mt < 4; ++mt) acc[1][mt] = wmma_bf16(af[mt], b1, acc[1][mt]);
        }
#pragma unroll
        for (int wt = 0; wt < 2; ++wt) {
            int n = (wave * 2 + wt) * 16 + (lane & 15);
            float bias = b_enc[n];
#pragma unroll
            for (int mt = 0; mt < 4; ++mt) {
#pragma unroll
                for (int r = 0; r < 8; ++r) {
                    int m = mt * 16 + r + (lane >> 4) * 8; // C/D: VGPR r -> rows r / r+8
                    float v = acc[wt][mt][r] + bias;
                    v = v > 0.0f ? v : 0.0f;
                    Xtile[m * 256 + n] = (__bf16)v;
                }
            }
        }
    }
    __syncthreads();

    // GEMM2: xg = x * W_ih^T + (b_ih + b_hh); wave owns 8 N-tiles, done in pairs
#pragma unroll
    for (int wtp = 0; wtp < 4; ++wtp) {
        f32x8 acc[2][4] = {};
        const int nb0 = (wave * 8 + wtp * 2 + 0) * 16;
        const int nb1 = (wave * 8 + wtp * 2 + 1) * 16;
#pragma unroll
        for (int s = 0; s < 8; ++s) {
            bf16x16 af[4];
#pragma unroll
            for (int mt = 0; mt < 4; ++mt)
                af[mt] = load_a_frag(Xtile + mt * 16 * 256, lane, s);
            bf16x16 b0 = load_b_frag(Wih, nb0, lane, s);
            bf16x16 b1 = load_b_frag(Wih, nb1, lane, s);
#pragma unroll
            for (int mt = 0; mt < 4; ++mt) acc[0][mt] = wmma_bf16(af[mt], b0, acc[0][mt]);
#pragma unroll
            for (int mt = 0; mt < 4; ++mt) acc[1][mt] = wmma_bf16(af[mt], b1, acc[1][mt]);
        }
#pragma unroll
        for (int wt = 0; wt < 2; ++wt) {
            int n = (wave * 8 + wtp * 2 + wt) * 16 + (lane & 15);
            float bias = bias_sum[n];
#pragma unroll
            for (int mt = 0; mt < 4; ++mt) {
#pragma unroll
                for (int r = 0; r < 8; ++r) {
                    int m = mt * 16 + r + (lane >> 4) * 8;
                    xg[(row0 + m) * GG + n] = acc[wt][mt][r] + bias;
                }
            }
        }
    }
}

// ---- K2: persistent LSTM scan ----------------------------------------------
// 4 workgroups (16 batch rows each), 32 waves; each wave owns 2 gate N-tiles
// with its W_hh slice resident in 128 VGPRs. h_prev A-fragments are hoisted
// into 64 VGPRs once per step and reused across both N-tiles. All per-step
// sync is an in-WGP s_barrier; h is exchanged through LDS only.
__global__ __launch_bounds__(1024, 1) void lstm_kernel(
    const __bf16* __restrict__ Whh, const float* __restrict__ xg,
    float* __restrict__ out) {
    extern __shared__ unsigned char smem[];
    __bf16* Htile = (__bf16*)smem;                  // 16x256 bf16 = 8 KB (h_prev)
    float*  Gbuf  = (float*)(smem + 8 * 1024);      // 16x1024 f32 = 64 KB (gates)

    const int tid  = threadIdx.x;
    const int lane = tid & 31;
    const int wave = tid >> 5;   // 0..31
    const int bg   = blockIdx.x; // batch group 0..3 -> rows bg*16..bg*16+15

    // Preload W_hh fragments into registers: 2 N-tiles x 8 K-steps
    bf16x16 wf[16];
#pragma unroll
    for (int wt = 0; wt < 2; ++wt) {
        int nb = (wave * 2 + wt) * 16;
#pragma unroll
        for (int s = 0; s < 8; ++s)
            wf[wt * 8 + s] = load_b_frag(Whh, nb, lane, s);
    }

    // h0 = 0, c0 = 0
    for (int i = tid; i < 16 * 256; i += 1024) Htile[i] = (__bf16)0.0f;
    float cst[4] = {0.0f, 0.0f, 0.0f, 0.0f};

    const int urow = tid >> 6;        // update: row 0..15
    const int uj   = (tid & 63) * 4;  // update: 4 consecutive hidden cols

    for (int t = 0; t < TT; ++t) {
        __syncthreads(); // Htile (h_prev) ready

        // Hoist h_prev fragments for this step (shared by both N-tiles)
        bf16x16 af[8];
#pragma unroll
        for (int s = 0; s < 8; ++s)
            af[s] = load_a_frag(Htile, lane, s);

        // gates = xg[:, t, :] + h_prev * W_hh^T
#pragma unroll
        for (int wt = 0; wt < 2; ++wt) {
            int nb = (wave * 2 + wt) * 16;
            int n  = nb + (lane & 15);
            f32x8 acc;
#pragma unroll
            for (int r = 0; r < 8; ++r) {
                int m = r + (lane >> 4) * 8;
                acc[r] = xg[(((long)(bg * 16 + m)) * TT + t) * GG + n];
            }
#pragma unroll
            for (int s = 0; s < 8; ++s)
                acc = wmma_bf16(af[s], wf[wt * 8 + s], acc);
#pragma unroll
            for (int r = 0; r < 8; ++r) {
                int m = r + (lane >> 4) * 8;
                Gbuf[m * 1024 + n] = acc[r];
            }
        }

        // Prefetch next steps' xg rows (global_prefetch_b8); 512 x 128B lines
        if (t + 8 < TT && tid < 512) {
            const float* pf =
                &xg[(((long)(bg * 16 + (tid >> 5))) * TT + (t + 8)) * GG + (tid & 31) * 32];
            __builtin_prefetch(pf, 0, 1);
        }

        __syncthreads(); // Gbuf complete; Htile free to overwrite

        // Elementwise cell update: thread owns (urow, uj..uj+3)
        float4 iv = *(const float4*)&Gbuf[urow * 1024 + uj];
        float4 fv = *(const float4*)&Gbuf[urow * 1024 + 256 + uj];
        float4 gv = *(const float4*)&Gbuf[urow * 1024 + 512 + uj];
        float4 ov = *(const float4*)&Gbuf[urow * 1024 + 768 + uj];
        float ia[4] = {iv.x, iv.y, iv.z, iv.w};
        float fa[4] = {fv.x, fv.y, fv.z, fv.w};
        float ga[4] = {gv.x, gv.y, gv.z, gv.w};
        float oa[4] = {ov.x, ov.y, ov.z, ov.w};
        float ha[4];
        bf16x4 hb;
#pragma unroll
        for (int u = 0; u < 4; ++u) {
            float ig = sigmf(ia[u]);
            float fg = sigmf(fa[u]);
            float gg = tanhf(ga[u]);
            float og = sigmf(oa[u]);
            cst[u] = fg * cst[u] + ig * gg;
            float hv = og * tanhf(cst[u]);
            ha[u] = hv;
            hb[u] = (__bf16)hv;
        }
        *(bf16x4*)&Htile[urow * 256 + uj] = hb; // feeds next step's A fragments
        *(float4*)&out[(((long)(bg * 16 + urow)) * TT + t) * HH + uj] =
            make_float4(ha[0], ha[1], ha[2], ha[3]);
    }
}

// ---- launcher ---------------------------------------------------------------
extern "C" void kernel_launch(void* const* d_in, const int* in_sizes, int n_in,
                              void* d_out, int out_size, void* d_ws, size_t ws_size,
                              hipStream_t stream) {
    const float* h    = (const float*)d_in[0];
    const float* Wenc = (const float*)d_in[1];
    const float* benc = (const float*)d_in[2];
    const float* Wih  = (const float*)d_in[3];
    const float* Whh  = (const float*)d_in[4];
    const float* bih  = (const float*)d_in[5];
    const float* bhh  = (const float*)d_in[6];
    float* out = (float*)d_out;

    // Workspace layout (bytes):
    //   [0, 128K)          W_enc bf16  (256x256)
    //   [128K, 640K)       W_ih  bf16  (1024x256)
    //   [640K, 1152K)      W_hh  bf16  (1024x256)
    //   [1152K, +4K)       bias_sum f32 (1024)
    //   [1183744, +512MB)  xg f32 (B*T*4H)
    unsigned char* ws = (unsigned char*)d_ws;
    unsigned short* wencb = (unsigned short*)(ws + 0);
    unsigned short* wihb  = (unsigned short*)(ws + 131072);
    unsigned short* whhb  = (unsigned short*)(ws + 655360);
    float* bias_sum       = (float*)(ws + 1179648);
    float* xg             = (float*)(ws + 1183744);

    prep_kernel<<<(GG * HH + 255) / 256, 256, 0, stream>>>(
        Wenc, Wih, Whh, bih, bhh, wencb, wihb, whhb, bias_sum);

    encoder_kernel<<<(BB * TT) / 64, 256, 64 * 1024, stream>>>(
        h, benc, (const __bf16*)wencb, (const __bf16*)wihb, bias_sum, xg);

    lstm_kernel<<<4, 1024, 8 * 1024 + 64 * 1024, stream>>>(
        (const __bf16*)whhb, xg, out);
}
